// AttentionMatcher_63050119905621
// MI455X (gfx1250) — matrix operand order repaired and checked by reading, not compiled, not run on hardware
//
#include <hip/hip_runtime.h>
#include <hip/hip_bf16.h>
#include <cmath>

typedef __attribute__((ext_vector_type(16))) __bf16 v16bf;
typedef __attribute__((ext_vector_type(8)))  float  v8f;
typedef unsigned int uint32;
typedef __attribute__((ext_vector_type(4))) unsigned int v4u;
typedef __attribute__((ext_vector_type(4))) int v4i;
typedef __attribute__((ext_vector_type(8))) int v8i;

#define NROWS 8192
#define EMBED 256
#define ROWS_PER_WAVE 16
#define WAVES_PER_WG  4
#define ROWS_PER_WG   (ROWS_PER_WAVE * WAVES_PER_WG)   // 64
#define KT            32                               // keys per tile
#define NKT           (NROWS / KT)                     // 256 key tiles
#define LOG2E 1.44269504088896340736f

#define FAST_EXP2(x) __builtin_amdgcn_exp2f(x)
#define FAST_RCP(x)  __builtin_amdgcn_rcpf(x)

// ---------------------------------------------------------------------------
// Tensor Data Mover descriptor builders (CDNA5 ISA ch.8, D# groups 0/1).
// 2D tensor, bf16 (data_size=1 -> 2 bytes), no multicast / iterate / pad.
// ---------------------------------------------------------------------------
__device__ __forceinline__ v4u tdm_g0(uint32 lds_addr, unsigned long long gaddr) {
    v4u g;
    g.x = 1u;                                            // count=1, user mode
    g.y = lds_addr;                                      // lds_addr [63:32]
    g.z = (uint32)gaddr;                                 // global_addr[31:0]
    g.w = (uint32)((gaddr >> 32) & 0x01FFFFFFull)        // global_addr[56:32]
        | (2u << 30);                                    // type=2 ("image")
    return g;
}

__device__ __forceinline__ v8i tdm_g1(uint32 tensor_d0, uint32 tensor_d1,
                                      uint32 tile_d0, uint32 tile_d1,
                                      unsigned long long stride0) {
    v8i g;
    g[0] = (int)(1u << 16);                              // data_size=1 (2 bytes)
    g[1] = (int)((tensor_d0 & 0xFFFFu) << 16);           // tensor_dim0[15:0] @ [63:48]
    g[2] = (int)((tensor_d0 >> 16) | ((tensor_d1 & 0xFFFFu) << 16)); // td0 hi | td1 lo
    g[3] = (int)((tensor_d1 >> 16) | (tile_d0 << 16));   // td1 hi | tile_dim0
    g[4] = (int)(tile_d1 & 0xFFFFu);                     // tile_dim1, tile_dim2=0
    g[5] = (int)(uint32)stride0;                         // tensor_dim0_stride[31:0]
    g[6] = (int)((uint32)(stride0 >> 32) & 0xFFFFu);     // stride0[47:32], dim1_stride=0
    g[7] = 0;
    return g;
}

__device__ __forceinline__ void tdm_load(v4u g0, v8i g1) {
    v4i z4 = {0, 0, 0, 0};
#if defined(__clang_major__) && (__clang_major__ >= 23)
    v8i z8 = {0, 0, 0, 0, 0, 0, 0, 0};
    __builtin_amdgcn_tensor_load_to_lds(g0, g1, z4, z4, z8, 0);
#else
    __builtin_amdgcn_tensor_load_to_lds(g0, g1, z4, z4, 0);
#endif
}

// ---------------------------------------------------------------------------
// Prep kernel: M (f32) -> workspace as bf16, row-major AND transposed, so the
// main loop's tile staging is two plain 2D DMAs (no per-tile convert/transpose).
// ---------------------------------------------------------------------------
__global__ __launch_bounds__(256)
void ConvertM_kernel(const float* __restrict__ Mp,
                     __bf16* __restrict__ Mbf,     // [NROWS][EMBED]
                     __bf16* __restrict__ Mtbf)    // [EMBED][NROWS]
{
    __shared__ __bf16 tile[KT][EMBED + 8];
    const int kb = blockIdx.x * KT;
    const int t  = threadIdx.x;
    const int r  = t >> 3;          // 0..31
    const int cb = (t & 7) * 32;    // 0..224

    const float* src  = Mp  + (size_t)(kb + r) * EMBED + cb;
    __bf16*      drow = Mbf + (size_t)(kb + r) * EMBED + cb;
    #pragma unroll
    for (int i = 0; i < 32; ++i) {
        __bf16 b = (__bf16)src[i];
        tile[r][cb + i] = b;
        drow[i] = b;
    }
    __syncthreads();

    // thread t == d writes Mt[d][kb..kb+31] (64B coalesced store)
    const int d = t;
    uint32 pack[16];
    #pragma unroll
    for (int i = 0; i < 16; ++i) {
        union { __bf16 h[2]; uint32 u; } cv;
        cv.h[0] = tile[2 * i + 0][d];
        cv.h[1] = tile[2 * i + 1][d];
        pack[i] = cv.u;
    }
    uint4* dst = (uint4*)(Mtbf + (size_t)d * NROWS + kb);
    #pragma unroll
    for (int i = 0; i < 4; ++i)
        dst[i] = make_uint4(pack[4 * i], pack[4 * i + 1], pack[4 * i + 2], pack[4 * i + 3]);
}

// ---------------------------------------------------------------------------
// Main flash-attention kernel: TDM double-buffered tiles + bf16 WMMA.
// Q fragments live in LDS (A-fragment layout) to stay under 256 VGPRs.
// ---------------------------------------------------------------------------
__global__ __launch_bounds__(ROWS_PER_WG * 2)  // 128 threads = 4 wave32
void AttentionMatcher_kernel(const __bf16* __restrict__ Mbf,
                             const __bf16* __restrict__ Mtbf,
                             const float* __restrict__ Np,
                             const float* __restrict__ gw,
                             const float* __restrict__ gwb,
                             const float* __restrict__ gb,
                             float* __restrict__ out)
{
    __shared__ __align__(32) __bf16 sMbuf [2][KT][EMBED];         // 2 x 16 KB (key-major)
    __shared__ __align__(32) __bf16 sMtbuf[2][EMBED][KT];         // 2 x 16 KB (d-major)
    __shared__ __align__(32) __bf16 sP [WAVES_PER_WG][16][KT];    // 4 KB P staging
    __shared__ __align__(32) __bf16 sQ [WAVES_PER_WG][8][32][16]; // 32 KB Q A-fragments

    const int tid  = threadIdx.x;
    const int lane = tid & 31;
    const int wave = tid >> 5;
    const int half = lane >> 4;
    const int l16  = lane & 15;
    const int qw   = blockIdx.x * ROWS_PER_WG + wave * ROWS_PER_WAVE;

    // Issue DMA of tile 0 into buffer 0 (one TDM pair per workgroup).
    if (wave == 0) {
        tdm_load(tdm_g0((uint32)(uintptr_t)&sMbuf[0][0][0],
                        (unsigned long long)(uintptr_t)Mbf),
                 tdm_g1(EMBED, NROWS, EMBED, KT, EMBED));
        tdm_load(tdm_g0((uint32)(uintptr_t)&sMtbuf[0][0][0],
                        (unsigned long long)(uintptr_t)Mtbf),
                 tdm_g1(NROWS, EMBED, KT, EMBED, NROWS));
    }

    // ---- Q = N[qw..qw+15, :] as bf16 A-fragments, staged into LDS per lane
    // (32 B per (chunk, lane), read back as 2x ds_load_b128 each iteration).
    // A 16x32, contiguous k order (same as B): lane<16 -> row=l16, K=c*32+0..15;
    //                        lane>=16 -> row=l16, K=c*32+16..31.
    {
        const float* qptr = Np + (size_t)(qw + l16) * EMBED;
        #pragma unroll
        for (int c = 0; c < 8; ++c) {
            const int k0 = c * 32 + half * 16;
            const int k1 = c * 32 + half * 16 + 8;
            v16bf qf;
            #pragma unroll
            for (int i = 0; i < 8; ++i) {
                qf[i]     = (__bf16)qptr[k0 + i];
                qf[8 + i] = (__bf16)qptr[k1 + i];
            }
            *(v16bf*)&sQ[wave][c][lane][0] = qf;
        }
    }
    asm volatile("s_wait_dscnt 0x0" ::: "memory");

    v8f O[16] = {};
    float mrow[8], lrow[8];
    #pragma unroll
    for (int j = 0; j < 8; ++j) { mrow[j] = -3.0e38f; lrow[j] = 0.0f; }

    for (int kt = 0; kt < NKT; ++kt) {
        const int buf = kt & 1;
        if (wave == 0) __builtin_amdgcn_s_wait_tensorcnt(0);
        __syncthreads();   // tile kt resident for all waves; prev compute done

        // Prefetch tile kt+1 into the other buffer while we compute on this one.
        if (wave == 0 && kt + 1 < NKT) {
            const unsigned long long koff = (unsigned long long)(kt + 1) * KT * 2ull;
            tdm_load(tdm_g0((uint32)(uintptr_t)&sMbuf[buf ^ 1][0][0],
                            (unsigned long long)(uintptr_t)Mbf + koff * EMBED),
                     tdm_g1(EMBED, NROWS, EMBED, KT, EMBED));
            tdm_load(tdm_g0((uint32)(uintptr_t)&sMtbuf[buf ^ 1][0][0],
                            (unsigned long long)(uintptr_t)Mtbf + koff),
                     tdm_g1(NROWS, EMBED, KT, EMBED, NROWS));
        }

        const __bf16 (*cM)[EMBED] = sMbuf[buf];
        const __bf16 (*cMt)[KT]   = sMtbuf[buf];
        const int kbase = kt * KT;

        // ---- S tiles: 16 WMMAs (A from sQ, B = 16 contiguous bf16 per lane).
        v8f S0 = {}, S1 = {};
        #pragma unroll
        for (int c = 0; c < 8; ++c) {
            v16bf qa = *(const v16bf*)&sQ[wave][c][lane][0];
            v16bf b0 = *(const v16bf*)&cM[l16][c * 32 + half * 16];
            S0 = __builtin_amdgcn_wmma_f32_16x16x32_bf16(false, qa, false, b0,
                                                         (short)0, S0, false, false);
            v16bf b1 = *(const v16bf*)&cM[16 + l16][c * 32 + half * 16];
            S1 = __builtin_amdgcn_wmma_f32_16x16x32_bf16(false, qa, false, b1,
                                                         (short)0, S1, false, false);
        }

        // ---- Zero the diagonal (reference sets it to 0.0, not -inf).
        if (kbase + KT > qw && kbase < qw + ROWS_PER_WAVE) {
            #pragma unroll
            for (int j = 0; j < 8; ++j) {
                const int q = qw + j + 8 * half;
                if (q == kbase + l16)      S0[j] = 0.0f;
                if (q == kbase + 16 + l16) S1[j] = 0.0f;
            }
        }

        // ---- Online softmax (rows live across 16-lane half groups).
        float pscale[8];
        int maxChanged = 0;
        #pragma unroll
        for (int j = 0; j < 8; ++j) {
            float t = fmaxf(S0[j], S1[j]);
            #pragma unroll
            for (int m = 8; m >= 1; m >>= 1) t = fmaxf(t, __shfl_xor(t, m, 32));
            const float mold = mrow[j];
            const float mnew = fmaxf(mold, t);
            if (mnew > mold) maxChanged = 1;
            const float sc = FAST_EXP2((mold - mnew) * LOG2E);
            mrow[j]   = mnew;
            pscale[j] = sc;
            const float p0 = FAST_EXP2((S0[j] - mnew) * LOG2E);
            const float p1 = FAST_EXP2((S1[j] - mnew) * LOG2E);
            S0[j] = p0; S1[j] = p1;
            float rs = p0 + p1;
            #pragma unroll
            for (int m = 8; m >= 1; m >>= 1) rs += __shfl_xor(rs, m, 32);
            lrow[j] = lrow[j] * sc + rs;
        }
        // Rescale O only when some row max actually moved (wave-uniform branch;
        // skipped in the common steady-state, saving 64 VALU ops/iteration).
        if (__any(maxChanged)) {
            #pragma unroll
            for (int t = 0; t < 16; ++t)
                #pragma unroll
                for (int j = 0; j < 8; ++j) O[t][j] *= pscale[j];
        }

        // ---- P (C layout) -> A-fragment via wave-private LDS bounce.
        #pragma unroll
        for (int j = 0; j < 8; ++j) {
            sP[wave][j + 8 * half][l16]      = (__bf16)S0[j];
            sP[wave][j + 8 * half][16 + l16] = (__bf16)S1[j];
        }
        asm volatile("s_wait_dscnt 0x0" ::: "memory");  // wave-local LDS RAW fence
        v16bf pfrag;
        #pragma unroll
        for (int i = 0; i < 8; ++i) {
            pfrag[i]     = sP[wave][l16][half * 16 + i];
            pfrag[8 + i] = sP[wave][l16][half * 16 + 8 + i];
        }

        // ---- O += P(16x32) x Mtile(32x16): 16 WMMAs from transposed LDS tile.
        #pragma unroll
        for (int t = 0; t < 16; ++t) {
            v16bf bo = *(const v16bf*)&cMt[t * 16 + l16][half * 16];
            O[t] = __builtin_amdgcn_wmma_f32_16x16x32_bf16(false, pfrag, false, bo,
                                                           (short)0, O[t], false, false);
        }
    }

    // ---- Normalize by softmax denominator.
    float rinv[8];
    #pragma unroll
    for (int j = 0; j < 8; ++j) rinv[j] = FAST_RCP(lrow[j]);
    #pragma unroll
    for (int t = 0; t < 16; ++t)
        #pragma unroll
        for (int j = 0; j < 8; ++j) O[t][j] *= rinv[j];

    // ---- Gate: sigmoid(O . gw + gwb + gb) per query row.
    float gpart[8] = {0.f, 0.f, 0.f, 0.f, 0.f, 0.f, 0.f, 0.f};
    #pragma unroll
    for (int t = 0; t < 16; ++t) {
        const float w = gw[t * 16 + l16];
        #pragma unroll
        for (int j = 0; j < 8; ++j) gpart[j] += O[t][j] * w;
    }
    const float bias = gwb[0] + gb[0];
    float gate[8];
    #pragma unroll
    for (int j = 0; j < 8; ++j) {
        float g = gpart[j];
        #pragma unroll
        for (int m = 8; m >= 1; m >>= 1) g += __shfl_xor(g, m, 32);
        g += bias;
        gate[j] = FAST_RCP(1.0f + FAST_EXP2(-g * LOG2E));   // sigmoid
    }

    // ---- boosted = O*g + N*(1-g); coalesced 16-lane stores per tile row.
    #pragma unroll
    for (int t = 0; t < 16; ++t) {
        #pragma unroll
        for (int j = 0; j < 8; ++j) {
            const int q = qw + j + 8 * half;
            const int d = t * 16 + l16;
            const float nv = Np[(size_t)q * EMBED + d];
            out[(size_t)q * EMBED + d] = O[t][j] * gate[j] + nv * (1.0f - gate[j]);
        }
    }
}

extern "C" void kernel_launch(void* const* d_in, const int* in_sizes, int n_in,
                              void* d_out, int out_size, void* d_ws, size_t ws_size,
                              hipStream_t stream) {
    (void)in_sizes; (void)n_in; (void)out_size; (void)ws_size;
    const float* M   = (const float*)d_in[0];
    const float* N   = (const float*)d_in[1];
    const float* gw  = (const float*)d_in[2];
    const float* gwb = (const float*)d_in[3];
    const float* gb  = (const float*)d_in[4];
    float* out = (float*)d_out;

    // Workspace: bf16 copies of M, row-major + transposed (4 MB each = 8 MB).
    __bf16* Mbf  = (__bf16*)d_ws;
    __bf16* Mtbf = (__bf16*)d_ws + (size_t)NROWS * EMBED;

    ConvertM_kernel<<<dim3(NROWS / KT), dim3(256), 0, stream>>>(M, Mbf, Mtbf);

    AttentionMatcher_kernel<<<dim3(NROWS / ROWS_PER_WG), dim3(ROWS_PER_WG * 2), 0, stream>>>(
        Mbf, Mtbf, N, gw, gwb, gb, out);
}
